// GeometricTransformerBlock_17454747091355
// MI455X (gfx1250) — matrix operand order repaired and checked by reading, not compiled, not run on hardware
//
#include <hip/hip_runtime.h>
#include <hip/hip_bf16.h>
#include <math.h>

#define C_HID   256
#define N_HEAD  8
#define D_HEAD  32
#define N_RBF   32
#define N_SEQ   512
#define N_BAT   2
#define F_HID   1024

typedef __attribute__((ext_vector_type(2)))  float  v2f;
typedef __attribute__((ext_vector_type(8)))  float  v8f;
typedef __attribute__((ext_vector_type(16))) __bf16 v16bf;

__device__ __forceinline__ float silu_f(float x) { return x / (1.f + __expf(-x)); }

// ---------------------------------------------------------------- LayerNorm
__global__ void ln_kernel(const float* __restrict__ X, const float* __restrict__ g,
                          const float* __restrict__ b, float* __restrict__ Y) {
  __shared__ float red[256];
  const int row = blockIdx.x;
  const int c   = threadIdx.x;
  float v = X[row * C_HID + c];
  red[c] = v; __syncthreads();
  for (int s = 128; s > 0; s >>= 1) { if (c < s) red[c] += red[c + s]; __syncthreads(); }
  float mu = red[0] * (1.f / C_HID); __syncthreads();
  float d = v - mu;
  red[c] = d * d; __syncthreads();
  for (int s = 128; s > 0; s >>= 1) { if (c < s) red[c] += red[c + s]; __syncthreads(); }
  float var = red[0] * (1.f / C_HID);
  Y[row * C_HID + c] = d * rsqrtf(var + 1e-5f) * g[c] + b[c];
}

// ------------------------------------------- Generic fp32 WMMA GEMM (one wave = 16x16 tile)
// Out[M,Nc] = act(A[M,K] @ W[K,Nc] + bias) + res
__global__ void gemm_f32_wmma(const float* __restrict__ A, const float* __restrict__ W,
                              const float* __restrict__ bias, const float* __restrict__ res,
                              float* __restrict__ Out, int M, int K, int Nc, int act) {
  const int lane = threadIdx.x & 31;
  const int hl = lane >> 4, l = lane & 15;
  const int tn = blockIdx.x, tm = blockIdx.y;
  const int arow = tm * 16 + l;   // A row handled by this lane
  const int col  = tn * 16 + l;   // B column handled by this lane
  const float* __restrict__ ap = A + (long)arow * K;
  v8f acc = {};
  for (int k0 = 0; k0 < K; k0 += 4) {
    const int ka = k0 + hl * 2;
    v2f a, bb;
    a.x  = ap[ka];             a.y  = ap[ka + 1];
    bb.x = W[(long)ka * Nc + col]; bb.y = W[(long)(ka + 1) * Nc + col];
    acc = __builtin_amdgcn_wmma_f32_16x16x4_f32(false, a, false, bb, (short)0, acc,
                                                false, false);
  }
  const float bv = bias ? bias[col] : 0.f;
  #pragma unroll
  for (int r = 0; r < 8; r++) {
    const int orow = tm * 16 + hl * 8 + r;
    float v = acc[r] + bv;
    if (act) v = silu_f(v);
    if (res) v += res[(long)orow * Nc + col];
    Out[(long)orow * Nc + col] = v;
  }
}

// -------------------------- Fused RBF expansion + bias MLP (all-WMMA, bf16)
// 2 waves / block, 32 pairs / block. GEMM1: K=32 (one WMMA per hidden tile).
// GEMM2: hidden(256) -> heads(8) also on WMMA via LDS re-layout, K=8x32.
// Writes geom bias laid out as (B,H,N,N).
__global__ void __launch_bounds__(64) bias_mlp_kernel(const float* __restrict__ dist,
        const float* __restrict__ w1, const float* __restrict__ b1,
        const float* __restrict__ w2, const float* __restrict__ b2,
        float* __restrict__ geom) {
  __shared__ float hsh[32 * C_HID];            // 32 KB: 32 pair-rows x 256 hidden (post-SiLU)
  const int tid  = threadIdx.x;
  const int wv   = tid >> 5, lane = tid & 31;
  const int hl   = lane >> 4, l = lane & 15;
  const int pbase = blockIdx.x * 32 + wv * 16;

  const float SIG  = 10.f / 32.f;              // SIGMA
  const float STEP = 10.f / 31.f;              // linspace(0,10,32) step
  const float d = dist[pbase + l];

  // A-matrix (16x32 bf16): lane(hl,l) holds K = hl*8+[0..7] then 16+hl*8+[0..7]
  v16bf a;
  #pragma unroll
  for (int e = 0; e < 16; e++) {
    const int kk = (e < 8) ? (hl * 8 + e) : (16 + hl * 8 + (e - 8));
    const float t = (d - kk * STEP) / SIG;
    a[e] = (__bf16)__expf(-t * t);
  }

  for (int t = 0; t < 16; t++) {               // 16 hidden-column tiles of 16
    const int n = t * 16 + l;
    v16bf bm;
    #pragma unroll
    for (int e = 0; e < 16; e++)               // B-matrix 32x16: lane(hl) holds K = hl*8+[0..7] then 16+hl*8+[0..7]
      bm[e] = (__bf16)w1[((e < 8) ? (hl * 8 + e) : (16 + hl * 8 + (e - 8))) * C_HID + n];
    const float bb = b1[n];
    v8f c;
    #pragma unroll
    for (int r = 0; r < 8; r++) c[r] = bb;
    c = __builtin_amdgcn_wmma_f32_16x16x32_bf16(false, a, false, bm, (short)0, c,
                                                false, false);
    #pragma unroll
    for (int r = 0; r < 8; r++) {
      const int rowg = wv * 16 + hl * 8 + r;
      hsh[rowg * C_HID + n] = silu_f(c[r]);
    }
  }
  __syncthreads();

  // ---- GEMM2 on WMMA: out(16 pairs x 8 heads) = silu_h(16x256) @ w2(256x8)
  // B is zero-padded to 16 columns; only cols 0..7 (heads) are stored.
  const float b2v = b2[l & 7];
  v8f acc2;
  #pragma unroll
  for (int r = 0; r < 8; r++) acc2[r] = b2v;
  const float* __restrict__ hrow = hsh + (wv * 16 + l) * C_HID;
  #pragma unroll
  for (int kc = 0; kc < 8; kc++) {             // K = 256 in chunks of 32
    const int kbase = kc * 32;
    v16bf a2, b2m;
    #pragma unroll
    for (int e = 0; e < 16; e++) {
      const int kk = kbase + ((e < 8) ? (hl * 8 + e) : (16 + hl * 8 + (e - 8)));
      a2[e] = (__bf16)hrow[kk];
    }
    #pragma unroll
    for (int e = 0; e < 16; e++) {
      const int kk = kbase + ((e < 8) ? (hl * 8 + e) : (16 + hl * 8 + (e - 8)));
      b2m[e] = (l < N_HEAD) ? (__bf16)w2[kk * N_HEAD + l] : (__bf16)0.f;
    }
    acc2 = __builtin_amdgcn_wmma_f32_16x16x32_bf16(false, a2, false, b2m, (short)0, acc2,
                                                   false, false);
  }
  if (l < N_HEAD) {
    #pragma unroll
    for (int r = 0; r < 8; r++) {
      const int pr = wv * 16 + hl * 8 + r;     // local pair row
      const int p  = blockIdx.x * 32 + pr;
      const int b  = p >> 18;                  // / (512*512)
      const int rem = p & (N_SEQ * N_SEQ - 1);
      const int i  = rem >> 9, j = rem & (N_SEQ - 1);
      geom[(((long)(b * N_HEAD + l) * N_SEQ) + i) * N_SEQ + j] = acc2[r];
    }
  }
}

// ------------------------------- scores = q.k^T/sqrt(D) + geom (in place), masked
__global__ void scores_kernel(const float* __restrict__ qkv,
                              const unsigned char* __restrict__ mask,
                              float* __restrict__ sc) {
  const int lane = threadIdx.x & 31;
  const int hl = lane >> 4, l = lane & 15;
  const int jt = blockIdx.x, it = blockIdx.y, bh = blockIdx.z;
  const int b = bh >> 3, h = bh & 7;
  const int i = it * 16 + l;
  const int j = jt * 16 + l;
  const float* __restrict__ qrow = qkv + (long)(b * N_SEQ + i) * (3 * C_HID) + h * D_HEAD;
  const float* __restrict__ krow = qkv + (long)(b * N_SEQ + j) * (3 * C_HID) + C_HID + h * D_HEAD;
  v8f acc = {};
  for (int k0 = 0; k0 < D_HEAD; k0 += 4) {
    const int ka = k0 + hl * 2;
    v2f a, bb;
    a.x  = qrow[ka]; a.y  = qrow[ka + 1];
    bb.x = krow[ka]; bb.y = krow[ka + 1];
    acc = __builtin_amdgcn_wmma_f32_16x16x4_f32(false, a, false, bb, (short)0, acc,
                                                false, false);
  }
  const float scale = 0.17677669529663687f;    // 1/sqrt(32)
  const bool mk = mask[b * N_SEQ + j] != 0;
  const long base = (long)bh * N_SEQ * N_SEQ;
  #pragma unroll
  for (int r = 0; r < 8; r++) {
    const int row = it * 16 + hl * 8 + r;
    const long idx = base + (long)row * N_SEQ + j;
    const float v = acc[r] * scale + sc[idx];
    sc[idx] = mk ? v : -1e9f;
  }
}

// ------------------------------------------------ row softmax over 512 (in place)
__global__ void softmax_kernel(float* __restrict__ sc) {
  __shared__ float red[256];
  float* __restrict__ p = sc + (long)blockIdx.x * N_SEQ;
  const int t = threadIdx.x;
  const float a0 = p[t], a1 = p[t + 256];
  red[t] = fmaxf(a0, a1); __syncthreads();
  for (int s = 128; s > 0; s >>= 1) { if (t < s) red[t] = fmaxf(red[t], red[t + s]); __syncthreads(); }
  const float mx = red[0]; __syncthreads();
  const float e0 = __expf(a0 - mx), e1 = __expf(a1 - mx);
  red[t] = e0 + e1; __syncthreads();
  for (int s = 128; s > 0; s >>= 1) { if (t < s) red[t] += red[t + s]; __syncthreads(); }
  const float inv = 1.f / red[0];
  p[t] = e0 * inv; p[t + 256] = e1 * inv;
}

// ------------------------------------------------ out[b,i,h*32+d] = attn @ v
__global__ void attnv_kernel(const float* __restrict__ attn, const float* __restrict__ qkv,
                             float* __restrict__ out) {
  const int lane = threadIdx.x & 31;
  const int hl = lane >> 4, l = lane & 15;
  const int dt = blockIdx.x, it = blockIdx.y, bh = blockIdx.z;
  const int b = bh >> 3, h = bh & 7;
  const int i = it * 16 + l;
  const int dcol = dt * 16 + l;
  const float* __restrict__ arow = attn + ((long)bh * N_SEQ + i) * N_SEQ;
  v8f acc = {};
  for (int k0 = 0; k0 < N_SEQ; k0 += 4) {
    const int ka = k0 + hl * 2;
    v2f a, bb;
    a.x = arow[ka]; a.y = arow[ka + 1];
    const float* __restrict__ vp =
        qkv + (long)(b * N_SEQ + ka) * (3 * C_HID) + 2 * C_HID + h * D_HEAD + dcol;
    bb.x = vp[0];
    bb.y = vp[3 * C_HID];
    acc = __builtin_amdgcn_wmma_f32_16x16x4_f32(false, a, false, bb, (short)0, acc,
                                                false, false);
  }
  #pragma unroll
  for (int r = 0; r < 8; r++) {
    const int row = it * 16 + hl * 8 + r;
    out[(long)(b * N_SEQ + row) * C_HID + h * D_HEAD + dcol] = acc[r];
  }
}

// ----------------------------------------------------------------------------
extern "C" void kernel_launch(void* const* d_in, const int* in_sizes, int n_in,
                              void* d_out, int out_size, void* d_ws, size_t ws_size,
                              hipStream_t stream) {
  const float* x     = (const float*)d_in[0];
  const float* dist  = (const float*)d_in[1];
  const unsigned char* mask = (const unsigned char*)d_in[2];
  const float* qkv_w = (const float*)d_in[3];
  const float* qkv_b = (const float*)d_in[4];
  const float* out_w = (const float*)d_in[5];
  const float* out_b = (const float*)d_in[6];
  const float* bw1   = (const float*)d_in[7];
  const float* bb1   = (const float*)d_in[8];
  const float* bw2   = (const float*)d_in[9];
  const float* bb2   = (const float*)d_in[10];
  const float* ln1g  = (const float*)d_in[11];
  const float* ln1b  = (const float*)d_in[12];
  const float* ln2g  = (const float*)d_in[13];
  const float* ln2b  = (const float*)d_in[14];
  const float* fw1   = (const float*)d_in[15];
  const float* fb1   = (const float*)d_in[16];
  const float* fw2   = (const float*)d_in[17];
  const float* fb2   = (const float*)d_in[18];

  float* ws   = (float*)d_ws;
  float* xn   = ws;                         // 1024*256
  float* qkv  = xn  + 1024 * 256;           // 1024*768
  float* sc   = qkv + 1024 * 768;           // 2*8*512*512  (geom bias -> scores -> attn)
  float* aout = sc  + (long)N_BAT * N_HEAD * N_SEQ * N_SEQ; // 1024*256
  float* x2   = aout + 1024 * 256;          // 1024*256
  float* x2n  = x2   + 1024 * 256;          // 1024*256
  float* ffh  = x2n  + 1024 * 256;          // 1024*1024

  const int rows = N_BAT * N_SEQ;           // 1024

  ln_kernel<<<rows, 256, 0, stream>>>(x, ln1g, ln1b, xn);

  gemm_f32_wmma<<<dim3(3 * C_HID / 16, rows / 16), 32, 0, stream>>>(
      xn, qkv_w, qkv_b, nullptr, qkv, rows, C_HID, 3 * C_HID, 0);

  bias_mlp_kernel<<<(N_BAT * N_SEQ * N_SEQ) / 32, 64, 0, stream>>>(
      dist, bw1, bb1, bw2, bb2, sc);

  scores_kernel<<<dim3(N_SEQ / 16, N_SEQ / 16, N_BAT * N_HEAD), 32, 0, stream>>>(
      qkv, mask, sc);

  softmax_kernel<<<N_BAT * N_HEAD * N_SEQ, 256, 0, stream>>>(sc);

  attnv_kernel<<<dim3(D_HEAD / 16, N_SEQ / 16, N_BAT * N_HEAD), 32, 0, stream>>>(
      sc, qkv, aout);

  gemm_f32_wmma<<<dim3(C_HID / 16, rows / 16), 32, 0, stream>>>(
      aout, out_w, out_b, x, x2, rows, C_HID, C_HID, 0);

  ln_kernel<<<rows, 256, 0, stream>>>(x2, ln2g, ln2b, x2n);

  gemm_f32_wmma<<<dim3(F_HID / 16, rows / 16), 32, 0, stream>>>(
      x2n, fw1, fb1, nullptr, ffh, rows, C_HID, F_HID, 1);

  gemm_f32_wmma<<<dim3(C_HID / 16, rows / 16), 32, 0, stream>>>(
      ffh, fw2, fb2, x2, (float*)d_out, rows, F_HID, C_HID, 0);
}
